// Attention_15075335209419
// MI455X (gfx1250) — matrix operand order repaired and checked by reading, not compiled, not run on hardware
//
#include <hip/hip_runtime.h>
#include <math.h>
#include <stdint.h>

// ---------------------------------------------------------------------------
// Problem constants (from the reference)
// ---------------------------------------------------------------------------
#define DIM      1024
#define NHEADS   16
#define HEADDIM  64
#define BATCH    4
#define SEQ      2048
#define ROWS     (BATCH * SEQ)          // 8192
#define QKV_N    (3 * DIM)              // 3072
#define SCALE_F  (0.125f)               // 1/sqrt(64)

// ---------------------------------------------------------------------------
// Types
// ---------------------------------------------------------------------------
typedef __attribute__((ext_vector_type(8)))  unsigned short u16x8;
typedef __attribute__((ext_vector_type(16))) __bf16         bf16x16;
typedef __attribute__((ext_vector_type(8)))  float          f32x8;
typedef __attribute__((ext_vector_type(4)))  unsigned int   u32x4;
typedef __attribute__((ext_vector_type(8)))  int            i32x8;
typedef __attribute__((ext_vector_type(4)))  int            i32x4;

static __device__ __forceinline__ unsigned short f2bf(float f) {
  unsigned int u = __float_as_uint(f);
  u += 0x7FFFu + ((u >> 16) & 1u);       // round-to-nearest-even
  return (unsigned short)(u >> 16);
}

static __device__ __forceinline__ u16x8 ld8(const unsigned short* p) {
  return *reinterpret_cast<const u16x8*>(p);
}

static __device__ __forceinline__ bf16x16 mk_frag(u16x8 lo, u16x8 hi) {
  union { struct { u16x8 lo, hi; } p; bf16x16 v; } u;
  u.p.lo = lo; u.p.hi = hi;
  return u.v;
}

static __device__ __forceinline__ f32x8 wmma_bf16(bf16x16 a, bf16x16 b, f32x8 c) {
  return __builtin_amdgcn_wmma_f32_16x16x32_bf16(
      /*neg_a=*/false, a, /*neg_b=*/false, b,
      /*c_mod=*/(short)0, c, /*reuse_a=*/false, /*reuse_b=*/false);
}

// A fragment (16x32): lane l -> row base+(l&15); half h=l>>4:
//   elems[0..7]  = src[row][kb + h*8 .. +7]
//   elems[8..15] = src[row][kb + 16 + h*8 .. +7]
static __device__ __forceinline__ bf16x16 load_a_frag(const unsigned short* rowptr,
                                                      int kb, int lh) {
  u16x8 lo = ld8(rowptr + kb + lh * 8);
  u16x8 hi = ld8(rowptr + kb + 16 + lh * 8);
  return mk_frag(lo, hi);
}

// B fragment (32x16) with B[k][n] stored as Wrow[n][k]:
//   lane l -> col n=base+(l&15); elems[0..7] = W[n][kb + (l>>4)*8 .. +7],
//   elems[8..15] = W[n][kb + 16 + (l>>4)*8 .. +7]
static __device__ __forceinline__ bf16x16 load_b_frag(const unsigned short* colptr,
                                                      int kb, int lh) {
  u16x8 lo = ld8(colptr + kb + lh * 8);
  u16x8 hi = ld8(colptr + kb + 16 + lh * 8);
  return mk_frag(lo, hi);
}

// ---------------------------------------------------------------------------
// Tensor Data Mover: 2D tile load (global -> LDS) with LDS padding.
// D# per CDNA5 ISA ch.8: group0 = {ctrl, lds_addr, gaddr_lo, gaddr_hi|type},
// group1 = {flags/pad, dims, tile dims, strides}. Padding: 4 DWORDs inserted
// every 32 DWORDs -> a 64-elem bf16 row lands with 72-elem LDS stride.
// ---------------------------------------------------------------------------
#define TDM_PAD_WORD0 ((1u << 16) /*data_size=2B*/ | (1u << 20) /*pad_enable*/ | \
                       (4u << 22) /*pad_interval: 32 DWORDs*/ |                  \
                       (3u << 25) /*pad_amount: 4 DWORDs*/)

static __device__ __forceinline__ void tdm_load_2d_padded(
    unsigned lds_off_bytes, const unsigned short* gptr,
    unsigned tensor_d0, unsigned tensor_d1,
    unsigned tile_d0, unsigned tile_d1, unsigned stride0_elems) {
  const unsigned long long ga = (unsigned long long)(uintptr_t)gptr;

  u32x4 g0;
  g0[0] = 1u;                                         // count=1 (valid), user mode
  g0[1] = lds_off_bytes;                              // lds_addr
  g0[2] = (unsigned)ga;                               // global_addr[31:0]
  g0[3] = (unsigned)((ga >> 32) & 0x1FFFFFFu) | (2u << 30);  // addr[56:32] | type=2

  i32x8 g1;
  g1[0] = (int)TDM_PAD_WORD0;                         // mask=0, data_size, padding
  g1[1] = (int)((tensor_d0 & 0xFFFFu) << 16);         // atomic_addr=0 | dim0.lo
  g1[2] = (int)(((tensor_d0 >> 16) & 0xFFFFu) | ((tensor_d1 & 0xFFFFu) << 16));
  g1[3] = (int)(((tensor_d1 >> 16) & 0xFFFFu) | (tile_d0 << 16));
  g1[4] = (int)(tile_d1 & 0xFFFFu);                   // tile_dim1 | tile_dim2=0
  g1[5] = (int)stride0_elems;                         // tensor_dim0_stride[31:0]
  g1[6] = 0;                                          // stride0.hi | stride1.lo
  g1[7] = 0;

  i32x4 gz4 = {0, 0, 0, 0};
#if __clang_major__ >= 23
  i32x8 gz8 = {0, 0, 0, 0, 0, 0, 0, 0};
  __builtin_amdgcn_tensor_load_to_lds(g0, g1, gz4, gz4, gz8, 0);
#else
  __builtin_amdgcn_tensor_load_to_lds(g0, g1, gz4, gz4, 0);
#endif
}

static __device__ __forceinline__ unsigned lds_offset_of(const void* p) {
  typedef __attribute__((address_space(3))) unsigned short lds_us;
  return (unsigned)(unsigned long long)(uintptr_t)(lds_us*)p;
}

// ---------------------------------------------------------------------------
// Kernel 0: fp32 -> bf16 (vectorized x4)
// ---------------------------------------------------------------------------
__global__ void k_f32_to_bf16(const float* __restrict__ src,
                              unsigned short* __restrict__ dst, int n4) {
  int i = blockIdx.x * blockDim.x + threadIdx.x;
  if (i >= n4) return;
  const float4 v = reinterpret_cast<const float4*>(src)[i];
  ushort4 o;
  o.x = f2bf(v.x); o.y = f2bf(v.y); o.z = f2bf(v.z); o.w = f2bf(v.w);
  reinterpret_cast<ushort4*>(dst)[i] = o;
}

// ---------------------------------------------------------------------------
// GEMM via TDM staging: C[m][n] = sum_k A[m][k]*W[n][k] + bias[n]
// Block tile 128x128, K-chunk 64. Wave 0 issues two TDM descriptor loads
// (A tile 128x64, W tile 128x64, both landing padded at stride 72), waits
// TENSORcnt, then all 8 waves (4m x 2n) run WMMA from LDS.
// ---------------------------------------------------------------------------
#define LDS_STRIDE 72   // 64 elems + TDM pad (4 DWORDs per 32 DWORDs)

template <int OUT_BF16>
__global__ __launch_bounds__(256)
void k_gemm_tdm(const unsigned short* __restrict__ A,
                const unsigned short* __restrict__ W,
                const float* __restrict__ bias,
                void* __restrict__ Cout, int M, int N, int K) {
  __shared__ unsigned short Asm[128][LDS_STRIDE];
  __shared__ unsigned short Wsm[128][LDS_STRIDE];

  const int tid  = threadIdx.x;
  const int wave = tid >> 5;
  const int lane = tid & 31;
  const int lr   = lane & 15;
  const int lh   = lane >> 4;
  const int wm   = wave >> 1;           // 0..3 -> 32-row strip
  const int wn   = wave & 1;            // 0..1 -> 64-col strip

  const int m0 = blockIdx.x * 128;
  const int n0 = blockIdx.y * 128;

  const unsigned ldsA = lds_offset_of(&Asm[0][0]);
  const unsigned ldsW = lds_offset_of(&Wsm[0][0]);

  f32x8 acc[2][4] = {};

  for (int kb = 0; kb < K; kb += 64) {
    if (wave == 0) {
      tdm_load_2d_padded(ldsA, A + (size_t)m0 * K + kb,
                         (unsigned)K, (unsigned)M, 64u, 128u, (unsigned)K);
      tdm_load_2d_padded(ldsW, W + (size_t)n0 * K + kb,
                         (unsigned)K, (unsigned)N, 64u, 128u, (unsigned)K);
      __builtin_amdgcn_s_wait_tensorcnt(0);
    }
    __syncthreads();   // publish staged tiles to all waves

#pragma unroll
    for (int kk = 0; kk < 64; kk += 32) {
      bf16x16 af[2];
#pragma unroll
      for (int i = 0; i < 2; ++i)
        af[i] = load_a_frag(&Asm[wm * 32 + i * 16 + lr][0], kk, lh);
#pragma unroll
      for (int j = 0; j < 4; ++j) {
        bf16x16 bf = load_b_frag(&Wsm[wn * 64 + j * 16 + lr][0], kk, lh);
#pragma unroll
        for (int i = 0; i < 2; ++i) acc[i][j] = wmma_bf16(af[i], bf, acc[i][j]);
      }
    }
    __syncthreads();   // tiles consumed; next TDM may overwrite
  }

  // Epilogue: C layout -> lane l holds col n=(l&15), rows m=(l>>4)*8 + r
#pragma unroll
  for (int i = 0; i < 2; ++i) {
#pragma unroll
    for (int j = 0; j < 4; ++j) {
      const int col = n0 + wn * 64 + j * 16 + lr;
      const float bv = bias[col];
#pragma unroll
      for (int r = 0; r < 8; ++r) {
        const int row = m0 + wm * 32 + i * 16 + lh * 8 + r;
        const float v = acc[i][j][r] + bv;
        if (OUT_BF16)
          reinterpret_cast<unsigned short*>(Cout)[(size_t)row * N + col] = f2bf(v);
        else
          reinterpret_cast<float*>(Cout)[(size_t)row * N + col] = v;
      }
    }
  }
}

// ---------------------------------------------------------------------------
// Flash attention: one block per (b*h, 64 query rows). 128 threads = 4 waves,
// each wave owns 16 query rows. Streaming softmax over key tiles of 64.
// qkv: [ROWS, 3072] bf16 (Q | K | V per head inside row). out: [ROWS, 1024] bf16.
// ---------------------------------------------------------------------------
__global__ __launch_bounds__(128)
void k_attention(const unsigned short* __restrict__ qkv,
                 unsigned short* __restrict__ outb) {
  __shared__ unsigned short VT[64][72];         // V transposed: VT[d][key]
  __shared__ unsigned short Pbuf[4][16][72];    // per-wave P tile (16 x 64)

  const int bh = blockIdx.x;
  const int b  = bh >> 4;
  const int h  = bh & 15;
  const int q0 = blockIdx.y * 64;

  const int wave = threadIdx.x >> 5;
  const int lane = threadIdx.x & 31;
  const int lr   = lane & 15;
  const int lh   = lane >> 4;

  const size_t rowbase = (size_t)b * SEQ;
  const int kcol = DIM + h * HEADDIM;
  const int vcol = 2 * DIM + h * HEADDIM;

  // Q A-fragments for this wave's 16 rows (constant across key loop)
  const unsigned short* qrow =
      qkv + (rowbase + q0 + wave * 16 + lr) * (size_t)QKV_N + h * HEADDIM;
  bf16x16 aq[2];
  aq[0] = load_a_frag(qrow, 0, lh);
  aq[1] = load_a_frag(qrow, 32, lh);

  float rmax[8], rsum[8];
  f32x8 accO[4] = {};
#pragma unroll
  for (int r = 0; r < 8; ++r) { rmax[r] = -INFINITY; rsum[r] = 0.0f; }

  for (int kt = 0; kt < SEQ; kt += 64) {
    __syncthreads();   // protect VT against readers from previous iteration

    // Stage V tile transposed: VT[d][key] = V[kt+key][d]
    for (int idx = threadIdx.x; idx < 64 * 64; idx += 128) {
      const int key = idx >> 6;
      const int d   = idx & 63;
      VT[d][key] = qkv[(rowbase + kt + key) * (size_t)QKV_N + vcol + d];
    }
    __syncthreads();

    // S = Q * K^T  (16 rows x 64 keys per wave)
    f32x8 s[4] = {};
#pragma unroll
    for (int j = 0; j < 4; ++j) {
      const unsigned short* kp =
          qkv + (rowbase + kt + j * 16 + lr) * (size_t)QKV_N + kcol;
      s[j] = wmma_bf16(aq[0], load_b_frag(kp, 0, lh), s[j]);
      s[j] = wmma_bf16(aq[1], load_b_frag(kp, 32, lh), s[j]);
    }
#pragma unroll
    for (int j = 0; j < 4; ++j) s[j] *= SCALE_F;

    // Online softmax (row m = lh*8 + r lives across the 16-lane half)
#pragma unroll
    for (int r = 0; r < 8; ++r) {
      float m = fmaxf(fmaxf(s[0][r], s[1][r]), fmaxf(s[2][r], s[3][r]));
#pragma unroll
      for (int mask = 1; mask < 16; mask <<= 1)
        m = fmaxf(m, __shfl_xor(m, mask, 32));
      const float nm   = fmaxf(rmax[r], m);
      const float corr = __expf(rmax[r] - nm);
      rmax[r] = nm;

      float lsum = 0.0f;
#pragma unroll
      for (int j = 0; j < 4; ++j) {
        const float p = __expf(s[j][r] - nm);
        s[j][r] = p;
        lsum += p;
      }
#pragma unroll
      for (int mask = 1; mask < 16; mask <<= 1)
        lsum += __shfl_xor(lsum, mask, 32);
      rsum[r] = rsum[r] * corr + lsum;
#pragma unroll
      for (int jd = 0; jd < 4; ++jd) accO[jd][r] *= corr;

#pragma unroll
      for (int j = 0; j < 4; ++j)
        Pbuf[wave][lh * 8 + r][j * 16 + lr] = f2bf(s[j][r]);
    }
    __syncthreads();

    // O += P * V   (A = P[16x64] from LDS, B = V via VT[d][key])
#pragma unroll
    for (int kb2 = 0; kb2 < 2; ++kb2) {
      const int kb = kb2 * 32;
      const bf16x16 ap = load_a_frag(&Pbuf[wave][lr][0], kb, lh);
#pragma unroll
      for (int jd = 0; jd < 4; ++jd) {
        const bf16x16 bv = load_b_frag(&VT[jd * 16 + lr][0], kb, lh);
        accO[jd] = wmma_bf16(ap, bv, accO[jd]);
      }
    }
  }

  // Normalize and store bf16 output [ROWS, DIM], col = h*64 + d
#pragma unroll
  for (int jd = 0; jd < 4; ++jd) {
#pragma unroll
    for (int r = 0; r < 8; ++r) {
      const int   row = q0 + wave * 16 + lh * 8 + r;
      const int   col = h * HEADDIM + jd * 16 + lr;
      const float v   = accO[jd][r] / rsum[r];
      outb[(rowbase + row) * (size_t)DIM + col] = f2bf(v);
    }
  }
}

// ---------------------------------------------------------------------------
// Launch
// ---------------------------------------------------------------------------
extern "C" void kernel_launch(void* const* d_in, const int* in_sizes, int n_in,
                              void* d_out, int out_size, void* d_ws, size_t ws_size,
                              hipStream_t stream) {
  const float* x      = (const float*)d_in[0];   // [ROWS, DIM]
  const float* w_qkv  = (const float*)d_in[1];   // [3*DIM, DIM]
  const float* b_qkv  = (const float*)d_in[2];   // [3*DIM]
  const float* w_proj = (const float*)d_in[3];   // [DIM, DIM]
  const float* b_proj = (const float*)d_in[4];   // [DIM]
  float* out = (float*)d_out;                    // [ROWS, DIM]

  char* ws = (char*)d_ws;
  unsigned short* xb    = (unsigned short*)(ws);                              // 16 MiB
  unsigned short* wqkvb = (unsigned short*)(ws + (size_t)16 * 1024 * 1024);   //  6 MiB
  unsigned short* wprjb = (unsigned short*)(ws + (size_t)22 * 1024 * 1024);   //  2 MiB
  unsigned short* qkvb  = (unsigned short*)(ws + (size_t)24 * 1024 * 1024);   // 48 MiB
  unsigned short* attnb = (unsigned short*)(ws + (size_t)72 * 1024 * 1024);   // 16 MiB

  // fp32 -> bf16 conversions
  {
    int n4;
    n4 = (ROWS * DIM) / 4;
    k_f32_to_bf16<<<(n4 + 255) / 256, 256, 0, stream>>>(x, xb, n4);
    n4 = (QKV_N * DIM) / 4;
    k_f32_to_bf16<<<(n4 + 255) / 256, 256, 0, stream>>>(w_qkv, wqkvb, n4);
    n4 = (DIM * DIM) / 4;
    k_f32_to_bf16<<<(n4 + 255) / 256, 256, 0, stream>>>(w_proj, wprjb, n4);
  }

  // QKV projection: [8192,1024] x [3072,1024]^T + bias -> bf16 [8192,3072]
  {
    dim3 grid(ROWS / 128, QKV_N / 128);
    k_gemm_tdm<1><<<grid, 256, 0, stream>>>(xb, wqkvb, b_qkv, qkvb,
                                            ROWS, QKV_N, DIM);
  }

  // Attention: one block per (b*h, 64 query rows)
  {
    dim3 grid(BATCH * NHEADS, SEQ / 64);
    k_attention<<<grid, 128, 0, stream>>>(qkvb, attnb);
  }

  // Output projection: [8192,1024] x [1024,1024]^T + bias -> f32 d_out
  {
    dim3 grid(ROWS / 128, DIM / 128);
    k_gemm_tdm<0><<<grid, 256, 0, stream>>>(attnb, wprjb, b_proj, out,
                                            ROWS, DIM, DIM);
  }
}
